// MHA_25744033972232
// MI455X (gfx1250) — matrix operand order repaired and checked by reading, not compiled, not run on hardware
//
#include <hip/hip_runtime.h>
#include <hip/hip_bf16.h>

// ---------------------------------------------------------------------------
// XCiT-style cross-covariance attention for MI455X (gfx1250, wave32, WMMA).
// Big projection GEMMs double-buffer their shared B-tiles through LDS using
// CDNA5 async global->LDS copies (ASYNCcnt); all matrix math is
// v_wmma_f32_16x16x32_f16 with per-lane fragment loads per the CDNA5 layout.
// ---------------------------------------------------------------------------

typedef _Float16 v16h __attribute__((ext_vector_type(16)));
typedef _Float16 v8h  __attribute__((ext_vector_type(8)));
typedef float    v8f  __attribute__((ext_vector_type(8)));

#define NQ   100
#define NT   197
#define BSZ  128
#define E    768
#define NH   12
#define HD   64
#define BH   (BSZ * NH)   // 1536
#define NQP  128          // NQ padded (K & N use)
#define NTP  224          // NT padded (multiple of 32)
#define NQP2 112          // NQ padded to N-tile multiple for out-map GEMM
#define EPS  1e-12f
#define BROW 40           // LDS B-tile row stride in f16 (32 data + 8 pad -> 80B)

// ---------------- device helpers ----------------

__device__ __forceinline__ v8f wz() {
    v8f z;
#pragma unroll
    for (int i = 0; i < 8; ++i) z[i] = 0.f;
    return z;
}

__device__ __forceinline__ v8f wmma32(v16h a, v16h b, v8f c) {
    // D = A(16x32 f16) * B(32x16 f16) + C(16x16 f32)
    return __builtin_amdgcn_wmma_f32_16x16x32_f16(false, a, false, b,
                                                  (short)0, c, false, false);
}

// A fragment (16x32, f16 source, row-major, lda elements).
// lane L: m = m0 + (L&15); element e maps K = (e/8)*16 + (L>>4)*8 + e%8.
__device__ __forceinline__ v16h loadA_f16(const _Float16* A, int lda,
                                          int m0, int k0, int lane) {
    const _Float16* p = A + (size_t)(m0 + (lane & 15)) * lda + k0 + ((lane >> 4) << 3);
    v8h lo = *(const v8h*)(p);
    v8h hi = *(const v8h*)(p + 16);
    v16h r;
#pragma unroll
    for (int i = 0; i < 8; ++i) { r[i] = lo[i]; r[i + 8] = hi[i]; }
    return r;
}

// A fragment from f32 source (converted to f16 in registers).
__device__ __forceinline__ v16h loadA_f32(const float* A, int lda,
                                          int m0, int k0, int lane) {
    const float* p = A + (size_t)(m0 + (lane & 15)) * lda + k0 + ((lane >> 4) << 3);
    v8f lo = *(const v8f*)(p);
    v8f hi = *(const v8f*)(p + 16);
    v16h r;
#pragma unroll
    for (int i = 0; i < 8; ++i) { r[i] = (_Float16)lo[i]; r[i + 8] = (_Float16)hi[i]; }
    return r;
}

// B fragment (32x16) from an n-major matrix Bt[n][k] (ldb elements per row).
// lane L: n = n0 + (L&15); element e maps K = (e/8)*16 + (L>>4)*8 + e%8.
__device__ __forceinline__ v16h loadB_nm(const _Float16* Bt, int ldb,
                                         int n0, int k0, int lane) {
    const _Float16* p = Bt + (size_t)(n0 + (lane & 15)) * ldb + k0 + ((lane >> 4) << 3);
    v8h lo = *(const v8h*)(p);
    v8h hi = *(const v8h*)(p + 16);
    v16h r;
#pragma unroll
    for (int i = 0; i < 8; ++i) { r[i] = lo[i]; r[i + 8] = hi[i]; }
    return r;
}

// B fragment from an LDS-staged n-major tile (row stride BROW f16, pad ->
// 20*n mod 64 bank walk, conflict-free across the 16 lanes of a half-wave).
__device__ __forceinline__ v16h loadB_lds(const _Float16* Bs, int n0, int lane) {
    const _Float16* p = Bs + (n0 + (lane & 15)) * BROW + ((lane >> 4) << 3);
    v8h lo = *(const v8h*)(p);
    v8h hi = *(const v8h*)(p + 16);
    v16h r;
#pragma unroll
    for (int i = 0; i < 8; ++i) { r[i] = lo[i]; r[i + 8] = hi[i]; }
    return r;
}

// Async global->LDS 16B/lane copy (CDNA5 TDM-lite path, tracked by ASYNCcnt).
__device__ __forceinline__ void async_copy_b128(unsigned lds_off,
                                                const _Float16* gsrc) {
    unsigned long long ga = (unsigned long long)(uintptr_t)gsrc;
    asm volatile("global_load_async_to_lds_b128 %0, %1, off"
                 :: "v"(lds_off), "v"(ga) : "memory");
}
__device__ __forceinline__ void wait_async0() {
    asm volatile("s_wait_asynccnt 0x0" ::: "memory");
}

// C layout: VGPR r -> M = m0 + (lane>>4)*8 + r, N = n0 + (lane&15).
__device__ __forceinline__ void storeC_f16(_Float16* C, int ldc, int m0, int n0,
                                           int lane, v8f acc) {
    int n  = n0 + (lane & 15);
    int mb = m0 + ((lane >> 4) << 3);
#pragma unroll
    for (int r = 0; r < 8; ++r) C[(size_t)(mb + r) * ldc + n] = (_Float16)acc[r];
}

__device__ __forceinline__ void storeC_f32(float* C, int ldc, int m0, int n0,
                                           int lane, v8f acc) {
    int n  = n0 + (lane & 15);
    int mb = m0 + ((lane >> 4) << 3);
#pragma unroll
    for (int r = 0; r < 8; ++r) C[(size_t)(mb + r) * ldc + n] = acc[r];
}

__device__ __forceinline__ float wsum(float v) {
#pragma unroll
    for (int off = 16; off > 0; off >>= 1) v += __shfl_xor(v, off, 32);
    return v;
}
__device__ __forceinline__ float wmaxr(float v) {
#pragma unroll
    for (int off = 16; off > 0; off >>= 1) v = fmaxf(v, __shfl_xor(v, off, 32));
    return v;
}

// ---------------- prep kernels ----------------

__global__ void k_prep_weights(const float* __restrict__ Win,
                               const float* __restrict__ Wout,
                               _Float16* __restrict__ W16,
                               _Float16* __restrict__ Wo16) {
    const int N1 = 3 * E * E, N2 = E * E;
    for (int i = blockIdx.x * blockDim.x + threadIdx.x; i < N1 + N2;
         i += gridDim.x * blockDim.x) {
        if (i < N1) W16[i]       = (_Float16)Win[i];
        else        Wo16[i - N1] = (_Float16)Wout[i - N1];
    }
}

// QMt[t][q] = l2norm-over-q(q_mapper[t]) padded to [NTP][NQP];
// XM [q][t] = l2norm-over-t(x_mapper[q]) padded to [NQP2][NTP].
__global__ __launch_bounds__(32) void k_prep_mappers(const float* __restrict__ qmap,
                                                     const float* __restrict__ xmap,
                                                     _Float16* __restrict__ QMt,
                                                     _Float16* __restrict__ XM) {
    int lane = threadIdx.x, bx = blockIdx.x;
    if (bx < NTP) {
        int t = bx;
        _Float16* row = QMt + (size_t)t * NQP;
        if (t < NT) {
            const float* src = qmap + (size_t)t * NQ;
            float ss = 0.f;
            for (int q = lane; q < NQ; q += 32) { float v = src[q]; ss += v * v; }
            ss = wsum(ss);
            float sc = 1.f / fmaxf(sqrtf(ss), EPS);
            for (int q = lane; q < NQP; q += 32)
                row[q] = (q < NQ) ? (_Float16)(src[q] * sc) : (_Float16)0.f;
        } else {
            for (int q = lane; q < NQP; q += 32) row[q] = (_Float16)0.f;
        }
    } else {
        int q = bx - NTP;
        _Float16* row = XM + (size_t)q * NTP;
        if (q < NQ) {
            const float* src = xmap + (size_t)q * NT;
            float ss = 0.f;
            for (int t = lane; t < NT; t += 32) { float v = src[t]; ss += v * v; }
            ss = wsum(ss);
            float sc = 1.f / fmaxf(sqrtf(ss), EPS);
            for (int t = lane; t < NTP; t += 32)
                row[t] = (t < NT) ? (_Float16)(src[t] * sc) : (_Float16)0.f;
        } else {
            for (int t = lane; t < NTP; t += 32) row[t] = (_Float16)0.f;
        }
    }
}

__global__ void k_zero_vt_pad(_Float16* __restrict__ Vt) {
    const int total = BH * (NTP - NT) * HD;
    for (int i = blockIdx.x * blockDim.x + threadIdx.x; i < total;
         i += gridDim.x * blockDim.x) {
        int bh  = i / ((NTP - NT) * HD);
        int rem = i % ((NTP - NT) * HD);
        int r = rem / HD, d = rem % HD;
        Vt[(size_t)bh * NTP * HD + (size_t)(NT + r) * HD + d] = (_Float16)0.f;
    }
}

// ---------------- in-projection GEMM (async-LDS double-buffered B) ----------
// C[m,n] = sum_k A[m,k] * W[sect*768+n, k] + bias ; scatter into head layout.
// MODE 0: q -> Qn[bh][d][tok] ; 1: k -> Kn[bh][d][tok] ; 2: v -> Vt[bh][tok][d]
template <int MODE>
__global__ __launch_bounds__(256) void k_gemm_inproj(const float* __restrict__ A,
                                                     const _Float16* __restrict__ W16,
                                                     const float* __restrict__ bias,
                                                     _Float16* __restrict__ dst) {
    __shared__ _Float16 Bs[2][64 * BROW];          // 2 x 5KB double buffer
    const int tid  = threadIdx.x;
    const int lane = tid & 31, wave = tid >> 5;
    const int m0  = blockIdx.x * 128 + wave * 16;  // 8 waves x 16 rows
    const int n0b = blockIdx.y * 64;
    const _Float16* Bsec = W16 + (size_t)MODE * E * E;  // rows already n-major

    // per-thread 16B async chunk: row n_c (0..63), chunk c_c (0..3)
    const int n_c = tid >> 2, c_c = tid & 3;
    const _Float16* gsrc = Bsec + (size_t)(n0b + n_c) * E + c_c * 8;
    const unsigned lds0 = (unsigned)(uintptr_t)&Bs[0][n_c * BROW + c_c * 8];
    const unsigned lds1 = (unsigned)(uintptr_t)&Bs[1][n_c * BROW + c_c * 8];

    v8f acc[4];
#pragma unroll
    for (int j = 0; j < 4; ++j) acc[j] = wz();

    async_copy_b128(lds0, gsrc);                   // prologue: stage k-step 0
    wait_async0();
    __syncthreads();

    const int STEPS = E / 32;                      // 24
    for (int i = 0; i < STEPS; ++i) {
        const int cur = i & 1;
        if (i + 1 < STEPS)                         // prefetch next tile (overlaps)
            async_copy_b128(cur ? lds0 : lds1, gsrc + (i + 1) * 32);
        v16h a = loadA_f32(A, E, m0, i * 32, lane);
        const _Float16* Bl = &Bs[cur][0];
#pragma unroll
        for (int j = 0; j < 4; ++j)
            acc[j] = wmma32(a, loadB_lds(Bl, j * 16, lane), acc[j]);
        if (i + 1 < STEPS) {
            wait_async0();                         // my copy of next tile landed
            __syncthreads();                       // everyone's landed + reads done
        }
    }

    const int hh = lane >> 4;
#pragma unroll
    for (int j = 0; j < 4; ++j) {
        int n = n0b + j * 16 + (lane & 15);
        int h = n >> 6, d = n & 63;
        float bv = bias[MODE * E + n];
#pragma unroll
        for (int r = 0; r < 8; ++r) {
            int m = m0 + hh * 8 + r;
            int tok = m >> 7, b = m & 127;         // rows are [token, batch]
            int bh = b * NH + h;
            float val = acc[j][r] + bv;
            if (MODE == 0)
                dst[(size_t)bh * HD * NQP + (size_t)d * NQP + tok] = (_Float16)val;
            else if (MODE == 1)
                dst[(size_t)bh * HD * NTP + (size_t)d * NTP + tok] = (_Float16)val;
            else
                dst[(size_t)bh * NTP * HD + (size_t)tok * HD + d] = (_Float16)val;
        }
    }
}

// ---------------- row L2 normalize (over tokens) ----------------

__global__ __launch_bounds__(32) void k_normalize(_Float16* __restrict__ buf,
                                                  int len, int ldr) {
    int lane = threadIdx.x;
    _Float16* row = buf + (size_t)blockIdx.x * ldr;
    float ss = 0.f;
    for (int c = lane; c < len; c += 32) { float v = (float)row[c]; ss += v * v; }
    ss = wsum(ss);
    float sc = 1.f / fmaxf(sqrtf(ss), EPS);
    for (int c = lane; c < ldr; c += 32) {
        float v = (c < len) ? (float)row[c] * sc : 0.f;
        row[c] = (_Float16)v;   // also zeroes the padded tail
    }
}

// ---------------- batched 64xNxK GEMMs (one wave per 16x16 tile) ----------------

__global__ __launch_bounds__(128) void k_bgemm_f16out(const _Float16* __restrict__ A,
                                                      long long sA, int lda,
                                                      const _Float16* __restrict__ B,
                                                      long long sB, int ldb, int K,
                                                      _Float16* __restrict__ C,
                                                      long long sC, int ldc) {
    const int lane = threadIdx.x & 31, wave = threadIdx.x >> 5;
    const long long bh = blockIdx.x;
    const _Float16* Ab = A + bh * sA;
    const _Float16* Bb = B + bh * sB;
    _Float16* Cb = C + bh * sC;
    const int m0 = wave * 16, n0 = blockIdx.y * 16;
    v8f acc = wz();
    for (int k0 = 0; k0 < K; k0 += 32)
        acc = wmma32(loadA_f16(Ab, lda, m0, k0, lane),
                     loadB_nm(Bb, ldb, n0, k0, lane), acc);
    storeC_f16(Cb, ldc, m0, n0, lane, acc);
}

__global__ __launch_bounds__(128) void k_bgemm_f32out(const _Float16* __restrict__ A,
                                                      long long sA, int lda,
                                                      const _Float16* __restrict__ B,
                                                      long long sB, int ldb, int K,
                                                      float* __restrict__ C,
                                                      long long sC, int ldc) {
    const int lane = threadIdx.x & 31, wave = threadIdx.x >> 5;
    const long long bh = blockIdx.x;
    const _Float16* Ab = A + bh * sA;
    const _Float16* Bb = B + bh * sB;
    float* Cb = C + bh * sC;
    const int m0 = wave * 16, n0 = blockIdx.y * 16;
    v8f acc = wz();
    for (int k0 = 0; k0 < K; k0 += 32)
        acc = wmma32(loadA_f16(Ab, lda, m0, k0, lane),
                     loadB_nm(Bb, ldb, n0, k0, lane), acc);
    storeC_f32(Cb, ldc, m0, n0, lane, acc);
}

// out-map: O[d,q] = sum_t X[d,t]*XM[q,t]; scatter to Y[(q*BSZ+b)][h*64+d] (f16)
__global__ __launch_bounds__(128) void k_bgemm_omap(const _Float16* __restrict__ X,
                                                    long long sA, int lda,
                                                    const _Float16* __restrict__ XM,
                                                    int ldb,
                                                    _Float16* __restrict__ Y) {
    const int lane = threadIdx.x & 31, wave = threadIdx.x >> 5;
    const long long bh = blockIdx.x;
    const int b = (int)(bh / NH), h = (int)(bh % NH);
    const _Float16* Ab = X + bh * sA;
    const int m0 = wave * 16, n0 = blockIdx.y * 16;
    v8f acc = wz();
    for (int k0 = 0; k0 < NTP; k0 += 32)
        acc = wmma32(loadA_f16(Ab, lda, m0, k0, lane),
                     loadB_nm(XM, ldb, n0, k0, lane), acc);
    int q  = n0 + (lane & 15);
    int mb = m0 + ((lane >> 4) << 3);
    if (q < NQ) {
#pragma unroll
        for (int r = 0; r < 8; ++r) {
            int d = mb + r;
            Y[((size_t)q * BSZ + b) * E + h * HD + d] = (_Float16)acc[r];
        }
    }
}

// ---------------- softmax over channel dim (64), with temperature ----------------

__global__ __launch_bounds__(32) void k_softmax(const float* __restrict__ attF,
                                                const float* __restrict__ temp,
                                                _Float16* __restrict__ attH) {
    int lane = threadIdx.x;
    size_t row = blockIdx.x;                 // row = bh*64 + d
    int h = (int)((row >> 6) % NH);
    float t = temp[h];
    const float* src = attF + row * 64;
    float v0 = src[lane] * t, v1 = src[lane + 32] * t;
    float m = wmaxr(fmaxf(v0, v1));
    float e0 = expf(v0 - m), e1 = expf(v1 - m);
    float inv = 1.f / wsum(e0 + e1);
    attH[row * 64 + lane]      = (_Float16)(e0 * inv);
    attH[row * 64 + lane + 32] = (_Float16)(e1 * inv);
}

// ---------------- out-projection GEMM (async-LDS double-buffered B) ----------

__global__ __launch_bounds__(256) void k_gemm_outproj(const _Float16* __restrict__ A,
                                                      const _Float16* __restrict__ Wo,
                                                      const float* __restrict__ bias,
                                                      float* __restrict__ out) {
    __shared__ _Float16 Bs[2][64 * BROW];
    const int tid  = threadIdx.x;
    const int lane = tid & 31, wave = tid >> 5;
    const int m0  = blockIdx.x * 128 + wave * 16;
    const int n0b = blockIdx.y * 64;

    const int n_c = tid >> 2, c_c = tid & 3;
    const _Float16* gsrc = Wo + (size_t)(n0b + n_c) * E + c_c * 8;
    const unsigned lds0 = (unsigned)(uintptr_t)&Bs[0][n_c * BROW + c_c * 8];
    const unsigned lds1 = (unsigned)(uintptr_t)&Bs[1][n_c * BROW + c_c * 8];

    v8f acc[4];
#pragma unroll
    for (int j = 0; j < 4; ++j) acc[j] = wz();

    async_copy_b128(lds0, gsrc);
    wait_async0();
    __syncthreads();

    const int STEPS = E / 32;
    for (int i = 0; i < STEPS; ++i) {
        const int cur = i & 1;
        if (i + 1 < STEPS)
            async_copy_b128(cur ? lds0 : lds1, gsrc + (i + 1) * 32);
        v16h a = loadA_f16(A, E, m0, i * 32, lane);
        const _Float16* Bl = &Bs[cur][0];
#pragma unroll
        for (int j = 0; j < 4; ++j)
            acc[j] = wmma32(a, loadB_lds(Bl, j * 16, lane), acc[j]);
        if (i + 1 < STEPS) {
            wait_async0();
            __syncthreads();
        }
    }

    const int hh = lane >> 4;
#pragma unroll
    for (int j = 0; j < 4; ++j) {
        int n = n0b + j * 16 + (lane & 15);
        float bv = bias[n];
#pragma unroll
        for (int r = 0; r < 8; ++r) {
            int m = m0 + hh * 8 + r;
            out[(size_t)m * E + n] = acc[j][r] + bv;
        }
    }
}

// ---------------- launcher ----------------

extern "C" void kernel_launch(void* const* d_in, const int* in_sizes, int n_in,
                              void* d_out, int out_size, void* d_ws, size_t ws_size,
                              hipStream_t stream) {
    (void)in_sizes; (void)n_in; (void)out_size; (void)ws_size;

    const float* query = (const float*)d_in[0];
    const float* key   = (const float*)d_in[1];
    const float* value = (const float*)d_in[2];
    const float* win   = (const float*)d_in[3];
    const float* bin   = (const float*)d_in[4];
    const float* wout  = (const float*)d_in[5];
    const float* bout  = (const float*)d_in[6];
    const float* temp  = (const float*)d_in[7];
    const float* qmap  = (const float*)d_in[8];
    const float* xmap  = (const float*)d_in[9];
    float* out = (float*)d_out;

    char* ws = (char*)d_ws;
    _Float16* W16  = (_Float16*)(ws + 0);            //  3,538,944 B
    _Float16* Wo16 = (_Float16*)(ws + 3538944);      //  1,179,648 B
    _Float16* QMt  = (_Float16*)(ws + 4718592);      //     57,344 B
    _Float16* XM   = (_Float16*)(ws + 4775936);      //     50,176 B
    _Float16* Qn   = (_Float16*)(ws + 4826112);      // 25,165,824 B
    _Float16* Kn   = (_Float16*)(ws + 29991936);     // 44,040,192 B
    _Float16* Vt   = (_Float16*)(ws + 74032128);     // 44,040,192 B
    _Float16* qm   = (_Float16*)(ws + 118072320);    // 44,040,192 B
    float*    attF = (float*)   (ws + 162112512);    // 25,165,824 B  (end ~187 MB)
    // dead-buffer aliases (stream order guarantees no hazard):
    _Float16* attH = Qn;   // softmax output reuses Qn (dead after qm GEMM)
    _Float16* X    = Kn;   // attn*V reuses Kn (dead after attn GEMM)
    _Float16* Y    = qm;   // pre-out-proj activations reuse qm (dead after attn GEMM)

    k_prep_weights<<<2048, 256, 0, stream>>>(win, wout, W16, Wo16);
    k_prep_mappers<<<NTP + NQP2, 32, 0, stream>>>(qmap, xmap, QMt, XM);

    k_gemm_inproj<0><<<dim3(NQ * BSZ / 128, E / 64), 256, 0, stream>>>(query, W16, bin, Qn);
    k_gemm_inproj<1><<<dim3(NT * BSZ / 128, E / 64), 256, 0, stream>>>(key,   W16, bin, Kn);
    k_gemm_inproj<2><<<dim3(NT * BSZ / 128, E / 64), 256, 0, stream>>>(value, W16, bin, Vt);
    k_zero_vt_pad<<<2048, 256, 0, stream>>>(Vt);

    k_normalize<<<BH * HD, 32, 0, stream>>>(Qn, NQ, NQP);
    k_normalize<<<BH * HD, 32, 0, stream>>>(Kn, NT, NTP);

    // qm[bh][d][t] = Qn(64x128) x QMt^T  (shared B, n-major)
    k_bgemm_f16out<<<dim3(BH, NTP / 16), 128, 0, stream>>>(
        Qn, (long long)HD * NQP, NQP, QMt, 0, NQP, NQP, qm, (long long)HD * NTP, NTP);
    // attn[bh][d][e] = qm(64x224) x Kn^T  (Kn is [e][t] == n-major)
    k_bgemm_f32out<<<dim3(BH, HD / 16), 128, 0, stream>>>(
        qm, (long long)HD * NTP, NTP, Kn, (long long)HD * NTP, NTP, NTP,
        attF, (long long)HD * HD, HD);
    k_softmax<<<BH * HD, 32, 0, stream>>>(attF, temp, attH);
    // X[bh][d][t] = attn(64x64) x V  (Vt is [t][e] == n-major)
    k_bgemm_f16out<<<dim3(BH, NTP / 16), 128, 0, stream>>>(
        attH, (long long)HD * HD, HD, Vt, (long long)NTP * HD, HD, HD,
        X, (long long)HD * NTP, NTP);
    // Y[(q,b)][h*64+d] = X(64x224) x XM^T  (shared B, n-major)
    k_bgemm_omap<<<dim3(BH, NQP2 / 16), 128, 0, stream>>>(
        X, (long long)HD * NTP, NTP, XM, NTP, Y);

    k_gemm_outproj<<<dim3(NQ * BSZ / 128, E / 64), 256, 0, stream>>>(Y, Wo16, bout, out);
}
